// LSTM_38027640439223
// MI455X (gfx1250) — matrix operand order repaired and checked by reading, not compiled, not run on hardware
//
#include <hip/hip_runtime.h>
#include <stdint.h>

typedef __attribute__((ext_vector_type(16))) __bf16 v16bf;
typedef __attribute__((ext_vector_type(8)))  float  v8f;

#define NWG 32
#define THREADS 256
static constexpr int Bb = 128, Tt = 1024, Hh = 512;

__device__ __forceinline__ unsigned short f2bf(float f) {
    union { float f; uint32_t u; } v; v.f = f;
    uint32_t u = v.u;
    uint32_t r = u + 0x7fffu + ((u >> 16) & 1u);   // round-to-nearest-even
    return (unsigned short)(r >> 16);
}

// ---------------------------------------------------------------------------
// Prep 1: zero the barrier counter and h-state buffer 0 (ws is poisoned 0xAA).
// ---------------------------------------------------------------------------
__global__ void zero_kernel(unsigned* counter, unsigned* hbuf0_u32) {
    int i = blockIdx.x * blockDim.x + threadIdx.x;
    if (i == 0) *counter = 0u;
    if (i < (Bb * Hh * 2) / 4) hbuf0_u32[i] = 0u;   // 128*512 bf16 = 32768 dwords
}

// ---------------------------------------------------------------------------
// Prep 2: swizzle recurrent weights (f32, [k][n] row-major) into bf16 WMMA
// B-fragment layout: [ct=128][kt=16][lane=32][16 bf16], gate-major columns.
// lane<16: N=lane, K=kt*32+0..15 ; lane>=16: N=lane-16, K=kt*32+16..31.
// ---------------------------------------------------------------------------
__global__ void swz_kernel(const float* __restrict__ Wg, const float* __restrict__ Wi,
                           const float* __restrict__ Wf, const float* __restrict__ Wo,
                           unsigned short* __restrict__ out) {
    int i = blockIdx.x * blockDim.x + threadIdx.x;    // 0 .. 2048*512-1
    if (i >= 2048 * 512) return;
    int e    = i & 15;
    int lane = (i >> 4) & 31;
    int kt   = (i >> 9) & 15;
    int ct   = i >> 13;            // 0..127 column tiles (gate-major)
    int g    = ct >> 5;            // gate
    int wg   = ct & 31;            // owning workgroup
    int col  = wg * 16 + (lane & 15);
    int k    = kt * 32 + ((lane & 16) ? 16 : 0) + e;
    const float* W = (g == 0) ? Wg : (g == 1) ? Wi : (g == 2) ? Wf : Wo;
    out[i] = f2bf(W[(size_t)k * Hh + col]);
}

// ---------------------------------------------------------------------------
// Main persistent LSTM scan. 32 WGs x 256 threads; each WG owns 16 hidden
// columns x 4 gates with its weight slab resident in 64 KB of LDS.
// ---------------------------------------------------------------------------
__global__ __launch_bounds__(THREADS, 1) void lstm_main(
    const float* __restrict__ x,
    const float* __restrict__ Wgx, const float* __restrict__ Wix,
    const float* __restrict__ Wfx, const float* __restrict__ Wox,
    const float* __restrict__ bg,  const float* __restrict__ bi,
    const float* __restrict__ bf_, const float* __restrict__ bo,
    const unsigned short* __restrict__ Wswz,
    unsigned short* __restrict__ h0buf, unsigned short* __restrict__ h1buf,
    float* __restrict__ hfin, unsigned* __restrict__ counter)
{
    __shared__ unsigned short wlds[4 * 16 * 32 * 16];   // 64 KB: [gate][kt][lane][16]

    const int wg   = blockIdx.x;
    const int tid  = threadIdx.x;
    const int wave = tid >> 5;
    const int lane = tid & 31;
    const int lo   = lane & 15;
    const int hi   = lane >> 4;

    // Stage this WG's 64 KB weight slab into LDS (read from HBM exactly once).
    for (int g = 0; g < 4; ++g) {
        const uint4* src = (const uint4*)(Wswz + (size_t)(g * 32 + wg) * 8192);
        uint4*       dst = (uint4*)(wlds + g * 8192);
        for (int j = tid; j < 1024; j += THREADS) dst[j] = src[j];
    }
    __syncthreads();

    const int n = wg * 16 + lo;               // this lane's hidden column
    const float wxv[4] = { Wgx[n], Wix[n], Wfx[n], Wox[n] };
    const float bbv[4] = { bg[n],  bi[n],  bf_[n], bo[n]  };
    const int brow_a = wave * 16 + lo;        // A-fragment row (M = lane&15)

    // Per-lane base for this lane's B fragments, in uint4 units.
    // wlds ushort index: g*8192 + (kt*32+lane)*16  ->  uint4: g*1024 + (kt*32+lane)*2
    const uint4* bbase = (const uint4*)wlds + (size_t)lane * 2;

    float c_[8];
    #pragma unroll
    for (int r = 0; r < 8; ++r) c_[r] = 0.f;

    union V16 { v16bf v; uint4 q[2]; };

    for (int t = 0; t < Tt; ++t) {
        const unsigned short* hc = (t & 1) ? h1buf : h0buf;
        unsigned short*       hn = (t & 1) ? h0buf : h1buf;

        v8f acc[4];
        #pragma unroll
        for (int g = 0; g < 4; ++g)
            #pragma unroll
            for (int e = 0; e < 8; ++e) acc[g][e] = 0.f;

        // A row base: h[b = wave*16 + (lane&15)][*], bf16 row of 512 (1 KB)
        const uint4* arow = (const uint4*)(hc + (size_t)brow_a * Hh);

        #pragma unroll 4
        for (int kt = 0; kt < 16; ++kt) {
            // Distinct register sets per gate -> one clause of 8 ds_load_b128,
            // a single wait, then 4 back-to-back WMMAs sharing the A fragment.
            V16 a, b0, b1, b2, b3;
            a.q[0] = arow[kt * 4 + 2 * hi];     // K chunk 0..7  / 16..23
            a.q[1] = arow[kt * 4 + 2 * hi + 1]; // K chunk 8..15 / 24..31
            const uint4* bp = bbase + (size_t)kt * 64;   // (kt*32 lanes)*2
            b0.q[0] = bp[0];        b0.q[1] = bp[1];
            b1.q[0] = bp[1024];     b1.q[1] = bp[1025];
            b2.q[0] = bp[2048];     b2.q[1] = bp[2049];
            b3.q[0] = bp[3072];     b3.q[1] = bp[3073];
            acc[0] = __builtin_amdgcn_wmma_f32_16x16x32_bf16(
                         false, a.v, false, b0.v, (short)0, acc[0], false, false);
            acc[1] = __builtin_amdgcn_wmma_f32_16x16x32_bf16(
                         false, a.v, false, b1.v, (short)0, acc[1], false, false);
            acc[2] = __builtin_amdgcn_wmma_f32_16x16x32_bf16(
                         false, a.v, false, b2.v, (short)0, acc[2], false, false);
            acc[3] = __builtin_amdgcn_wmma_f32_16x16x32_bf16(
                         false, a.v, false, b3.v, (short)0, acc[3], false, false);
        }

        // Elementwise LSTM update. C/D layout: VGPR r -> M=r (lane<16) / r+8.
        #pragma unroll
        for (int r = 0; r < 8; ++r) {
            int   b  = wave * 16 + r + hi * 8;
            float xv = x[(size_t)b * Tt + t];
            float zg = acc[0][r] + xv * wxv[0] + bbv[0];
            float zi = acc[1][r] + xv * wxv[1] + bbv[1];
            float zf = acc[2][r] + xv * wxv[2] + bbv[2];
            float zo = acc[3][r] + xv * wxv[3] + bbv[3];
            float gg = tanhf(zg);
            float ii = 1.f / (1.f + __expf(-zi));
            float ff = 1.f / (1.f + __expf(-zf));
            float oo = 1.f / (1.f + __expf(-zo));
            float cv = gg * ii + c_[r] * ff;
            c_[r] = cv;
            float hv = tanhf(cv) * oo;
            hn[(size_t)b * Hh + n] = f2bf(hv);
            if (t == Tt - 1) hfin[(size_t)b * Hh + n] = hv;
        }

        // ---- grid barrier (monotonic counter; reset by zero_kernel per launch)
        __threadfence();
        __syncthreads();
        if (tid == 0) {
            __hip_atomic_fetch_add(counter, 1u, __ATOMIC_RELEASE, __HIP_MEMORY_SCOPE_AGENT);
            unsigned target = (unsigned)(t + 1) * NWG;
            while (__hip_atomic_load(counter, __ATOMIC_ACQUIRE, __HIP_MEMORY_SCOPE_AGENT) < target)
                __builtin_amdgcn_s_sleep(2);
        }
        __syncthreads();
        __builtin_amdgcn_fence(__ATOMIC_ACQUIRE, "agent");
    }
}

// ---------------------------------------------------------------------------
// Final projection: out[b,c] = h_T[b,:] . W_ph[c,:] + b_p[c]   (128 x 10)
// ---------------------------------------------------------------------------
__global__ void proj_kernel(const float* __restrict__ hfin, const float* __restrict__ Wph,
                            const float* __restrict__ bp, float* __restrict__ out) {
    int i = blockIdx.x * blockDim.x + threadIdx.x;
    if (i >= Bb * 10) return;
    int b = i / 10, c = i % 10;
    float s = bp[c];
    const float* hr = hfin + (size_t)b * Hh;
    const float* wr = Wph  + (size_t)c * Hh;
    for (int k = 0; k < Hh; ++k) s += hr[k] * wr[k];
    out[i] = s;
}

// ---------------------------------------------------------------------------
extern "C" void kernel_launch(void* const* d_in, const int* in_sizes, int n_in,
                              void* d_out, int out_size, void* d_ws, size_t ws_size,
                              hipStream_t stream) {
    const float* x   = (const float*)d_in[0];
    const float* Wgx = (const float*)d_in[1];
    const float* Wgh = (const float*)d_in[2];
    const float* bg  = (const float*)d_in[3];
    const float* Wix = (const float*)d_in[4];
    const float* Wih = (const float*)d_in[5];
    const float* bi  = (const float*)d_in[6];
    const float* Wfx = (const float*)d_in[7];
    const float* Wfh = (const float*)d_in[8];
    const float* bf  = (const float*)d_in[9];
    const float* Wox = (const float*)d_in[10];
    const float* Woh = (const float*)d_in[11];
    const float* bo  = (const float*)d_in[12];
    const float* Wph = (const float*)d_in[13];
    const float* bp  = (const float*)d_in[14];

    char* ws = (char*)d_ws;
    unsigned*       counter = (unsigned*)ws;                              // 256 B slot
    unsigned short* h0      = (unsigned short*)(ws + 256);                // 128 KB
    unsigned short* h1      = (unsigned short*)(ws + 256 + 131072);       // 128 KB
    unsigned short* wswz    = (unsigned short*)(ws + 256 + 262144);       // 2 MB
    float*          hfin    = (float*)(ws + 256 + 262144 + 2097152);      // 256 KB

    zero_kernel<<<128, 256, 0, stream>>>(counter, (unsigned*)h0);
    swz_kernel <<<4096, 256, 0, stream>>>(Wgh, Wih, Wfh, Woh, wswz);
    lstm_main  <<<NWG, THREADS, 0, stream>>>(x, Wgx, Wix, Wfx, Wox,
                                             bg, bi, bf, bo,
                                             wswz, h0, h1, hfin, counter);
    proj_kernel<<<5, 256, 0, stream>>>(hfin, Wph, bp, (float*)d_out);
}
